// MSDeformAttn_54992761258330
// MI455X (gfx1250) — matrix operand order repaired and checked by reading, not compiled, not run on hardware
//
#include <hip/hip_runtime.h>

typedef __attribute__((ext_vector_type(16))) __bf16 v16bf;
typedef __attribute__((ext_vector_type(8)))  __bf16 v8bf;
typedef __attribute__((ext_vector_type(8)))  float  v8f;

#define KDIM 256
#define LDSPAD 8
#define BSTRIDE (KDIM + LDSPAD)   // bf16 elements per LDS column-row (padded vs bank conflicts)
#define NPANEL 64                 // columns per block panel (4 N-tiles per wave)

// C[M x N] = A[M x 256] * B[256 x N] + bias[N]; bf16 WMMA, f32 accumulate.
// One wave32 computes a 16x64 strip: per k-step, build the A fragment once and
// load all 4 B fragments up-front so the 4 WMMAs issue back-to-back.
__global__ __launch_bounds__(256) void gemm_bf16_wmma(
    const float* __restrict__ A, const float* __restrict__ B,
    const float* __restrict__ bias, float* __restrict__ C,
    int M, int N)
{
  __shared__ __bf16 Blds[NPANEL * BSTRIDE];
  const int tid = threadIdx.x;
  const int tn0 = blockIdx.y * NPANEL;   // first column of this block's panel

  // Cooperative: Blds[n][k] = bf16(B[k][tn0+n])  (transposed: fragments are k-contiguous)
  for (int idx = tid; idx < NPANEL * KDIM; idx += 256) {
    int k  = idx >> 6;          // 0..255
    int nn = idx & (NPANEL - 1);
    Blds[nn * BSTRIDE + k] = (__bf16)B[(size_t)k * N + tn0 + nn];
  }
  __syncthreads();

  const int wave = tid >> 5;
  const int lane = tid & 31;
  const int tm   = blockIdx.x * 8 + wave;
  if (tm * 16 >= M) return;          // wave-uniform: EXEC stays all-ones for WMMA

  const int rowA   = tm * 16 + (lane & 15);
  const int kbaseA = (lane >> 4) << 3;   // 0 or 8  (A 16x32 bf16 layout)
  const int colB   = lane & 15;
  const int kbaseB = (lane >> 4) << 3;   // 0 or 8  (B 32x16 bf16 layout, same k order as A)

  v8f acc[4];
  #pragma unroll
  for (int j = 0; j < 4; ++j) {
    float bv = bias[tn0 + j * 16 + colB];  // bias depends on column only
    #pragma unroll
    for (int i = 0; i < 8; ++i) acc[j][i] = bv;
  }

  const float*  arow  = A + (size_t)rowA * KDIM;
  const __bf16* bbase = Blds + colB * BSTRIDE + kbaseB;

  #pragma unroll
  for (int k0 = 0; k0 < KDIM; k0 += 32) {
    // A fragment: lane holds K = kbaseA..+7 and K = 16+kbaseA..+7 (f32 -> bf16 in regs)
    v16bf a;
    const float* ap = arow + k0 + kbaseA;
    #pragma unroll
    for (int e = 0; e < 8; ++e) a[e] = (__bf16)ap[e];
    #pragma unroll
    for (int e = 0; e < 8; ++e) a[8 + e] = (__bf16)ap[16 + e];

    // Load all 4 B fragments first (8 independent ds_load_b128 -> one clause),
    // then issue the 4 WMMAs back-to-back.
    v16bf bfrag[4];
    #pragma unroll
    for (int j = 0; j < 4; ++j) {
      const __bf16* bp = bbase + j * 16 * BSTRIDE + k0;
      v8bf blo = *(const v8bf*)(bp);
      v8bf bhi = *(const v8bf*)(bp + 16);
      #pragma unroll
      for (int e = 0; e < 8; ++e) { bfrag[j][e] = blo[e]; bfrag[j][8 + e] = bhi[e]; }
    }

    #pragma unroll
    for (int j = 0; j < 4; ++j)
      acc[j] = __builtin_amdgcn_wmma_f32_16x16x32_bf16(false, a, false, bfrag[j],
                                                       (short)0, acc[j], false, false);
  }

  const int rbase = tm * 16 + ((lane >> 4) << 3);
  #pragma unroll
  for (int j = 0; j < 4; ++j) {
    const int col = tn0 + j * 16 + colB;
    #pragma unroll
    for (int i = 0; i < 8; ++i)
      C[(size_t)(rbase + i) * N + col] = acc[j][i];
  }
}

// One wave32 per (n, t1, q, head); lane = channel within head (d = 32).
// Softmax over 16 (level,point) logits; att is identical across frames -> fold 1/K.
__global__ __launch_bounds__(256) void msdeform_sample(
    const float* __restrict__ value,   // [(NB*T2*8500) x 256], channel = head*32+c
    const float* __restrict__ logits,  // [(NB*T1*8500) x 128], channel = head*16+(l*4+p)
    const float* __restrict__ offv,    // [(NB*T1*8500) x 256], channel = head*32+(l*4+p)*2+xy
    const float* __restrict__ refp,    // [NB,T1,8500,4,2]
    const int*   __restrict__ shapes,  // [4][2] = (h,w)
    const int*   __restrict__ lstart,  // [4]
    float* __restrict__ sampled)       // [(NB*T1*8500) x 256]
{
  const int wid  = blockIdx.x * 8 + (threadIdx.x >> 5);
  const int lane = threadIdx.x & 31;
  int idx = wid;
  const int head = idx & 7;  idx >>= 3;
  const int q    = idx % 8500; idx /= 8500;
  const int t1   = idx & 3;
  const int n    = idx >> 2;
  if (n >= 2) return;

  const size_t row = (size_t)(n * 4 + t1) * 8500 + q;

  // softmax over 16 logits for this head (broadcast over K frames -> divide by K)
  const float* lg = logits + row * 128 + head * 16;
  float att[16];
  float mx = -1e30f;
  #pragma unroll
  for (int i = 0; i < 16; ++i) { att[i] = lg[i]; mx = fmaxf(mx, att[i]); }
  float s = 0.f;
  #pragma unroll
  for (int i = 0; i < 16; ++i) { att[i] = __expf(att[i] - mx); s += att[i]; }
  const int t2lo = (t1 > 0) ? t1 - 1 : 0;
  const int t2hi = (t1 < 3) ? t1 + 1 : 3;
  const float invs = 1.f / (s * (float)(t2hi - t2lo + 1));
  #pragma unroll
  for (int i = 0; i < 16; ++i) att[i] *= invs;

  const float* ofs = offv + row * 256 + head * 32;
  const float* rp  = refp + row * 8;
  float acc = 0.f;

  for (int l = 0; l < 4; ++l) {
    const int H  = shapes[2 * l];
    const int W  = shapes[2 * l + 1];
    const int ls = lstart[l];
    const float refx = rp[2 * l];
    const float refy = rp[2 * l + 1];
    #pragma unroll
    for (int p = 0; p < 4; ++p) {
      const float a = att[l * 4 + p];
      // loc = ref + off/[W,H];  x_img = loc.x*W - 0.5 = ref.x*W + off.x - 0.5
      const float x = refx * (float)W + ofs[(l * 4 + p) * 2]     - 0.5f;
      const float y = refy * (float)H + ofs[(l * 4 + p) * 2 + 1] - 0.5f;
      const float x0f = floorf(x), y0f = floorf(y);
      const int   x0 = (int)x0f,  y0 = (int)y0f;
      const float fx = x - x0f,   fy = y - y0f;
      const bool vx0 = (x0 >= 0) && (x0 < W);
      const bool vx1 = (x0 >= -1) && (x0 + 1 < W);
      const bool vy0 = (y0 >= 0) && (y0 < H);
      const bool vy1 = (y0 >= -1) && (y0 + 1 < H);
      const float w00 = (vx0 && vy0) ? (1.f - fx) * (1.f - fy) : 0.f;
      const float w10 = (vx1 && vy0) ? fx * (1.f - fy)         : 0.f;
      const float w01 = (vx0 && vy1) ? (1.f - fx) * fy         : 0.f;
      const float w11 = (vx1 && vy1) ? fx * fy                 : 0.f;
      const int cx0 = min(max(x0,     0), W - 1);
      const int cx1 = min(max(x0 + 1, 0), W - 1);
      const int cy0 = min(max(y0,     0), H - 1);
      const int cy1 = min(max(y0 + 1, 0), H - 1);
      const size_t i00 = (size_t)(cy0 * W + cx0) * 256;
      const size_t i10 = (size_t)(cy0 * W + cx1) * 256;
      const size_t i01 = (size_t)(cy1 * W + cx0) * 256;
      const size_t i11 = (size_t)(cy1 * W + cx1) * 256;
      float ssum = 0.f;
      for (int t2 = t2lo; t2 <= t2hi; ++t2) {
        const float* vb = value + ((size_t)(n * 4 + t2) * 8500 + ls) * 256 + head * 32 + lane;
        ssum += w00 * vb[i00] + w10 * vb[i10] + w01 * vb[i01] + w11 * vb[i11];
      }
      acc += a * ssum;
    }
  }
  sampled[row * 256 + head * 32 + lane] = acc;
}

extern "C" void kernel_launch(void* const* d_in, const int* in_sizes, int n_in,
                              void* d_out, int out_size, void* d_ws, size_t ws_size,
                              hipStream_t stream)
{
  const float* query = (const float*)d_in[0];
  const float* refp  = (const float*)d_in[1];
  const float* inpf  = (const float*)d_in[2];
  const int*   shp   = (const int*)d_in[3];
  const int*   lst   = (const int*)d_in[4];
  const float* W_off = (const float*)d_in[5];
  const float* b_off = (const float*)d_in[6];
  const float* W_att = (const float*)d_in[7];
  const float* b_att = (const float*)d_in[8];
  const float* W_val = (const float*)d_in[9];
  const float* b_val = (const float*)d_in[10];
  const float* W_out = (const float*)d_in[11];
  const float* b_out = (const float*)d_in[12];

  const int Mrows = 2 * 4 * 8500;  // 68000 == NB*T1*LEN_Q == NB*T2*Len_in

  // workspace: value (66.4MB) | logits (33.2MB) | off (66.4MB) | sampled (66.4MB)
  float* value   = (float*)d_ws;
  float* logitsb = value   + (size_t)Mrows * 256;
  float* offv    = logitsb + (size_t)Mrows * 128;
  float* sampl   = offv    + (size_t)Mrows * 256;

  dim3 blk(256);
  const int mtiles = Mrows / 16;       // 4250
  const int gx     = (mtiles + 7) / 8; // 532 blocks of 8 wave-strips

  gemm_bf16_wmma<<<dim3(gx, 4), blk, 0, stream>>>(inpf,  W_val, b_val, value,   Mrows, 256);
  gemm_bf16_wmma<<<dim3(gx, 2), blk, 0, stream>>>(query, W_att, b_att, logitsb, Mrows, 128);
  gemm_bf16_wmma<<<dim3(gx, 4), blk, 0, stream>>>(query, W_off, b_off, offv,    Mrows, 256);

  const int nwaves = 2 * 4 * 8500 * 8; // 544000 -> 68000 blocks of 8 waves
  msdeform_sample<<<dim3(nwaves / 8), blk, 0, stream>>>(value, logitsb, offv, refp, shp, lst, sampl);

  gemm_bf16_wmma<<<dim3(gx, 4), blk, 0, stream>>>(sampl, W_out, b_out, (float*)d_out, Mrows, 256);
}
